// Head_1400159338985
// MI455X (gfx1250) — matrix operand order repaired and checked by reading, not compiled, not run on hardware
//
#include <hip/hip_runtime.h>
#include <hip/hip_bf16.h>

#define B_ 4
#define T_ 4096
#define C_ 1024
#define H_ 128

typedef __attribute__((ext_vector_type(16))) __bf16          v16bf;
typedef __attribute__((ext_vector_type(8)))  float           v8f;
typedef __attribute__((ext_vector_type(16))) unsigned short  v16u;

#if __has_builtin(__builtin_amdgcn_global_load_async_to_lds_b128)
#define HAVE_ASYNC 1
// exact parameter types per hipcc diagnostic: 'int __vector(4) __device__ *'
typedef int v4i_gcc __attribute__((vector_size(16)));
typedef __attribute__((address_space(1))) v4i_gcc* gvp_t;
typedef __attribute__((address_space(3))) v4i_gcc* lvp_t;
#endif

// fp32 -> bf16 round-to-nearest-even
__device__ __forceinline__ unsigned short f2bf(float f) {
    unsigned u = __builtin_bit_cast(unsigned, f);
    u += 0x7FFFu + ((u >> 16) & 1u);
    return (unsigned short)(u >> 16);
}
__device__ __forceinline__ unsigned pack2(float x, float y) {
    return (unsigned)f2bf(x) | ((unsigned)f2bf(y) << 16);
}

// A-matrix (16x32, 16-bit) element->K mapping per CDNA5 ISA 7.12.2
__device__ __forceinline__ int amap(int e, int g) {
    int p = e >> 1;
    int k = (p < 4) ? (2 * p + 8 * g) : (16 + 2 * (p - 4) + 8 * g);
    return k + (e & 1);
}

__device__ __forceinline__ v8f wmma_bf16(v16bf a, v16bf b, v8f c) {
    return __builtin_amdgcn_wmma_f32_16x16x32_bf16(
        false, a, false, b, (short)0, c, false, false);
}

// 16-byte global -> LDS copy: async direct-to-LDS when available
__device__ __forceinline__ void copy16(const void* gsrc, void* ldst) {
#ifdef HAVE_ASYNC
    __builtin_amdgcn_global_load_async_to_lds_b128(
        (gvp_t)(const __attribute__((address_space(1))) void*)gsrc,
        (lvp_t)(__attribute__((address_space(3))) void*)ldst, 0, 0);
#else
    *(uint4*)ldst = *(const uint4*)gsrc;
#endif
}
__device__ __forceinline__ void wait_copies() {
#ifdef HAVE_ASYNC
#if __has_builtin(__builtin_amdgcn_s_wait_asynccnt)
    __builtin_amdgcn_s_wait_asynccnt(0);
#else
    asm volatile("s_wait_asynccnt 0" ::: "memory");
#endif
#endif
}

// ---------------------------------------------------------------------------
// Kernel 1: projections. q written TRANSPOSED qT[b][h][t]; k PRE-SCALED by
// C^-0.5 (only used for scores); k,v row-major. 128 threads = 4 waves.
// ---------------------------------------------------------------------------
__global__ __launch_bounds__(128) void proj_kernel(
    const float* __restrict__ ix, const float* __restrict__ Wq,
    const float* __restrict__ Wk, const float* __restrict__ Wv,
    unsigned short* __restrict__ qkv)
{
    __shared__ unsigned short xs[64][40];    // ix tile bf16 (padded rows)
    __shared__ unsigned short ws[32][136];   // W  tile bf16 (padded rows)
    __shared__ unsigned short tb[128][64];   // transpose staging (q only)

    const int tid  = threadIdx.x;
    const int lane = tid & 31;
    const int w    = tid >> 5;
    const int g    = lane >> 4;
    const int ln   = lane & 15;

    const int m0  = blockIdx.x * 64;         // row block over B*T
    const int sel = blockIdx.y;              // 0=q 1=k 2=v
    const float* W = (sel == 0) ? Wq : (sel == 1) ? Wk : Wv;
    unsigned short* outp = qkv + (size_t)sel * (size_t)(B_ * T_) * H_;

    v8f acc[8];
#pragma unroll
    for (int i = 0; i < 8; ++i) acc[i] = (v8f)0.0f;

    for (int kc = 0; kc < C_ / 32; ++kc) {
        for (int i = tid; i < 512; i += 128) {      // ix tile: 64 x 8 float4
            int r = i >> 3, j = i & 7;
            float4 f = ((const float4*)(ix + (size_t)(m0 + r) * C_ + kc * 32))[j];
            ((unsigned*)&xs[r][0])[j * 2 + 0] = pack2(f.x, f.y);
            ((unsigned*)&xs[r][0])[j * 2 + 1] = pack2(f.z, f.w);
        }
        for (int i = tid; i < 1024; i += 128) {     // W tile: 32 x 32 float4
            int r = i >> 5, j = i & 31;
            float4 f = ((const float4*)(W + (size_t)(kc * 32 + r) * H_))[j];
            ((unsigned*)&ws[r][0])[j * 2 + 0] = pack2(f.x, f.y);
            ((unsigned*)&ws[r][0])[j * 2 + 1] = pack2(f.z, f.w);
        }
        __syncthreads();

        v16u au;
#pragma unroll
        for (int e = 0; e < 16; ++e) au[e] = xs[w * 16 + ln][amap(e, g)];
        v16bf a = __builtin_bit_cast(v16bf, au);

#pragma unroll
        for (int nt = 0; nt < 8; ++nt) {
            v16u bu;                                // B: K = element (table order), N = lane%16
#pragma unroll
            for (int e = 0; e < 16; ++e) bu[e] = ws[amap(e, g)][nt * 16 + ln];
            acc[nt] = wmma_bf16(a, __builtin_bit_cast(v16bf, bu), acc[nt]);
        }
        __syncthreads();
    }

    if (sel == 0) {
        // stage transposed, then write qT[b][h][t0..t0+63]
#pragma unroll
        for (int nt = 0; nt < 8; ++nt)
#pragma unroll
            for (int r = 0; r < 8; ++r)
                tb[nt * 16 + ln][w * 16 + r + 8 * g] = f2bf(acc[nt][r]);
        __syncthreads();
        const int b = m0 >> 12, tl = m0 & (T_ - 1);
        const int h = tid;
        uint4* dst = (uint4*)(outp + ((size_t)(b * H_ + h)) * T_ + tl);
        const uint4* src = (const uint4*)&tb[h][0];
#pragma unroll
        for (int j = 0; j < 8; ++j) dst[j] = src[j];
    } else {
        const float sc = (sel == 1) ? 0.03125f : 1.0f;   // C^-0.5 into k
#pragma unroll
        for (int nt = 0; nt < 8; ++nt)
#pragma unroll
            for (int r = 0; r < 8; ++r) {
                int m = m0 + w * 16 + r + 8 * g;
                outp[(size_t)m * H_ + nt * 16 + ln] = f2bf(acc[nt][r] * sc);
            }
    }
}

// ---------------------------------------------------------------------------
// Kernel 2: flash attention, double-buffered async s-tiles.
// score[t,s] = k[t].q[s] (k pre-scaled), causal s<=t. Block = 64 t-rows.
// ---------------------------------------------------------------------------
__global__ __launch_bounds__(128) void attn_kernel(
    const unsigned short* __restrict__ qT,
    const unsigned short* __restrict__ kb,
    const unsigned short* __restrict__ vb,
    float* __restrict__ out)
{
    __shared__ unsigned short ks[64][136];       // k strip (padded)
    __shared__ unsigned short qt[2][128][40];    // q^T tiles [h][s] (padded)
    __shared__ unsigned short vs[2][32][136];    // v tiles (padded)
    __shared__ unsigned short ps[4][16][40];     // per-wave P relayout

    const int tid  = threadIdx.x;
    const int lane = tid & 31;
    const int w    = tid >> 5;
    const int g    = lane >> 4;
    const int ln   = lane & 15;
    const int r0   = tid >> 4;       // coop-copy row base
    const int jj   = tid & 15;       // coop-copy chunk

    const int t0    = blockIdx.x * 64;
    const int b     = blockIdx.y;
    const int tbase = t0 + w * 16;
    const size_t bo = (size_t)b * T_ * H_;
    const unsigned short* qrow = qT + ((size_t)(b * H_ + tid)) * T_;

    // k strip [64][128] + first s-tile, one wait+barrier
#pragma unroll
    for (int u = 0; u < 8; ++u)
        copy16(kb + bo + (size_t)(t0 + r0 + 8 * u) * H_ + jj * 8,
               &ks[r0 + 8 * u][jj * 8]);
#pragma unroll
    for (int j = 0; j < 4; ++j)
        copy16(qrow + 0 + j * 8, &qt[0][tid][j * 8]);
#pragma unroll
    for (int u = 0; u < 4; ++u)
        copy16(vb + bo + (size_t)(0 + r0 + 8 * u) * H_ + jj * 8,
               &vs[0][r0 + 8 * u][jj * 8]);
    wait_copies();
    __syncthreads();

    v16bf akf[4];
#pragma unroll
    for (int hc = 0; hc < 4; ++hc) {
        v16u u;
#pragma unroll
        for (int e = 0; e < 16; ++e) u[e] = ks[w * 16 + ln][hc * 32 + amap(e, g)];
        akf[hc] = __builtin_bit_cast(v16bf, u);
    }

    v8f o[8];
#pragma unroll
    for (int i = 0; i < 8; ++i) o[i] = (v8f)0.0f;
    float mrow[8], lrow[8];
#pragma unroll
    for (int r = 0; r < 8; ++r) { mrow[r] = -1e30f; lrow[r] = 0.0f; }

    const int ntile = t0 / 32 + 2;               // s-tiles this block needs

    for (int it = 0; it < ntile; ++it) {
        const int s0  = it * 32;
        const int cur = it & 1;

        // issue next tile's async copies into the other buffer
        if (it + 1 < ntile) {
            const int nb = cur ^ 1, sn = s0 + 32;
#pragma unroll
            for (int j = 0; j < 4; ++j)
                copy16(qrow + sn + j * 8, &qt[nb][tid][j * 8]);
#pragma unroll
            for (int u = 0; u < 4; ++u)
                copy16(vb + bo + (size_t)(sn + r0 + 8 * u) * H_ + jj * 8,
                       &vs[nb][r0 + 8 * u][jj * 8]);
        }

        if (s0 < tbase + 16) {   // wave has unmasked columns in this tile
            v8f sacc[2];
            sacc[0] = (v8f)0.0f; sacc[1] = (v8f)0.0f;
#pragma unroll
            for (int hc = 0; hc < 4; ++hc)
#pragma unroll
                for (int nt = 0; nt < 2; ++nt) {
                    v16u bu;   // B: K = element (h, table order), N = lane%16 (s)
#pragma unroll
                    for (int e = 0; e < 16; ++e)
                        bu[e] = qt[cur][hc * 32 + amap(e, g)][nt * 16 + ln];
                    sacc[nt] = wmma_bf16(akf[hc], __builtin_bit_cast(v16bf, bu), sacc[nt]);
                }

            float p0[8], p1[8];
#pragma unroll
            for (int r = 0; r < 8; ++r) {
                int t = tbase + r + 8 * g;
                p0[r] = (s0 + ln      <= t) ? sacc[0][r] : -1e30f;
                p1[r] = (s0 + 16 + ln <= t) ? sacc[1][r] : -1e30f;
            }

            v8f cv;
#pragma unroll
            for (int r = 0; r < 8; ++r) {
                float cm = fmaxf(p0[r], p1[r]);
#pragma unroll
                for (int msk = 1; msk < 16; msk <<= 1)
                    cm = fmaxf(cm, __shfl_xor(cm, msk, 32));
                float mnew = fmaxf(mrow[r], cm);
                float corr = __expf(mrow[r] - mnew);
                float e0 = __expf(p0[r] - mnew);
                float e1 = __expf(p1[r] - mnew);
                float rs = e0 + e1;
#pragma unroll
                for (int msk = 1; msk < 16; msk <<= 1)
                    rs += __shfl_xor(rs, msk, 32);
                lrow[r] = lrow[r] * corr + rs;
                mrow[r] = mnew;
                cv[r] = corr;
                ps[w][r + 8 * g][ln]      = f2bf(e0);
                ps[w][r + 8 * g][16 + ln] = f2bf(e1);
            }
#pragma unroll
            for (int ot = 0; ot < 8; ++ot) o[ot] *= cv;   // v_pk_mul_f32

            v16u pu;
#pragma unroll
            for (int e = 0; e < 16; ++e) pu[e] = ps[w][ln][amap(e, g)];
            v16bf pf = __builtin_bit_cast(v16bf, pu);

#pragma unroll
            for (int ot = 0; ot < 8; ++ot) {
                v16u bu;   // B: K = element (s, table order), N = lane%16 (h)
#pragma unroll
                for (int e = 0; e < 16; ++e) bu[e] = vs[cur][amap(e, g)][ot * 16 + ln];
                o[ot] = wmma_bf16(pf, __builtin_bit_cast(v16bf, bu), o[ot]);
            }
        }

        wait_copies();       // next buffer landed while we computed
        __syncthreads();     // one barrier per iteration
    }

    v8f li;
#pragma unroll
    for (int r = 0; r < 8; ++r) li[r] = 1.0f / lrow[r];
#pragma unroll
    for (int ot = 0; ot < 8; ++ot) {
        v8f ov = o[ot] * li;
#pragma unroll
        for (int r = 0; r < 8; ++r) {
            int t = tbase + r + 8 * g;
            out[bo + (size_t)t * H_ + ot * 16 + ln] = ov[r];
        }
    }
}

extern "C" void kernel_launch(void* const* d_in, const int* in_sizes, int n_in,
                              void* d_out, int out_size, void* d_ws, size_t ws_size,
                              hipStream_t stream) {
    const float* ix = (const float*)d_in[0];
    const float* Wq = (const float*)d_in[1];
    const float* Wk = (const float*)d_in[2];
    const float* Wv = (const float*)d_in[3];
    float* out = (float*)d_out;

    unsigned short* qkv = (unsigned short*)d_ws;   // qT | k(prescaled) | v
    const size_t one = (size_t)(B_ * T_) * H_;

    dim3 g1(B_ * T_ / 64, 3);
    proj_kernel<<<g1, 128, 0, stream>>>(ix, Wq, Wk, Wv, qkv);

    dim3 g2(T_ / 64, B_);
    attn_kernel<<<g2, 128, 0, stream>>>(qkv, qkv + one, qkv + 2 * one, out);
}
